// CQAttention_19928648254003
// MI455X (gfx1250) — matrix operand order repaired and checked by reading, not compiled, not run on hardware
//
#include <hip/hip_runtime.h>
#include <math.h>

// ---------------------------------------------------------------------------
// CQAttention for MI455X (gfx1250): bf16 WMMA flash-attention.
//   Qp  = gelu(Q @ W^T + b)                  (kernel 1, also writes Qp^T)
//   out = softmax(C @ Qp^T, axis=q) @ Qp + C (kernel 2, online softmax,
//         64-q chunks, async double-buffered Qp staging in LDS)
// ---------------------------------------------------------------------------

typedef __attribute__((ext_vector_type(16))) __bf16 v16bf;
typedef __attribute__((ext_vector_type(8)))  __bf16 v8bf;
typedef __attribute__((ext_vector_type(8)))  float  v8f;
typedef __attribute__((ext_vector_type(4)))  float  v4f;

#define B_  8
#define QN  2048
#define DN  8192
#define H_  256
#define CH  64                  // q-chunk per iteration
#define QB_PITCH 264            // staged Qp row pitch (elems): 256 + 8 pad (bank spread)
#define LP_PITCH 72             // P-tile row pitch (elems): 64 + 8 pad

#if __has_builtin(__builtin_amdgcn_global_load_async_to_lds_b128)
#define HAS_ASYNC 1
// Builtin signature (from hipcc diagnostic): async b128 copy takes
//   (v4i addrspace(1)*, v4i addrspace(3)*, imm offset, imm cpol)
typedef int v4i_vs __attribute__((vector_size(16)));
typedef __attribute__((address_space(1))) v4i_vs gas_v4i;
typedef __attribute__((address_space(3))) v4i_vs las_v4i;
#else
#define HAS_ASYNC 0
#endif

__device__ __forceinline__ void wait_async_all() {
#if __has_builtin(__builtin_amdgcn_s_wait_asynccnt)
  __builtin_amdgcn_s_wait_asynccnt(0);
#elif HAS_ASYNC
  asm volatile("s_wait_asynccnt 0x0" ::: "memory");
#endif
}

__device__ __forceinline__ v8f wmma_bf16(v16bf a, v16bf b, v8f c) {
  return __builtin_amdgcn_wmma_f32_16x16x32_bf16(
      /*neg_a=*/false, a, /*neg_b=*/false, b,
      /*c_mod=*/(short)0, c, /*reuse_a=*/false, /*reuse_b=*/false);
}

// A-fragment (16x32, 16-bit): per-lane K = {off..off+7, off+16..off+23},
// caller passes p = row_base + kbase + (half*8).
__device__ __forceinline__ v16bf frag_a_from_f32(const float* p) {
  v4f a0 = *(const v4f*)(p + 0);
  v4f a1 = *(const v4f*)(p + 4);
  v4f b0 = *(const v4f*)(p + 16);
  v4f b1 = *(const v4f*)(p + 20);
  v16bf f;
#pragma unroll
  for (int i = 0; i < 4; ++i) {
    f[i]      = (__bf16)a0[i];
    f[4 + i]  = (__bf16)a1[i];
    f[8 + i]  = (__bf16)b0[i];
    f[12 + i] = (__bf16)b1[i];
  }
  return f;
}

__device__ __forceinline__ v16bf frag_a_from_bf16(const __bf16* p) {
  v8bf lo = *(const v8bf*)(p + 0);
  v8bf hi = *(const v8bf*)(p + 16);
  v16bf f;
#pragma unroll
  for (int i = 0; i < 8; ++i) { f[i] = lo[i]; f[8 + i] = hi[i]; }
  return f;
}

// B-fragment (32x16, 16-bit): per-lane K = {off..off+7, off+16..off+23},
// caller passes p = col_base + kbase + (half*8).
__device__ __forceinline__ v16bf frag_b_from_bf16(const __bf16* p) {
  v8bf lo = *(const v8bf*)(p + 0);
  v8bf hi = *(const v8bf*)(p + 16);
  v16bf f;
#pragma unroll
  for (int i = 0; i < 8; ++i) { f[i] = lo[i]; f[8 + i] = hi[i]; }
  return f;
}

__device__ __forceinline__ v16bf frag_b_from_f32(const float* p) {
  v16bf f;
#pragma unroll
  for (int j = 0; j < 4; ++j) {
    v4f v = *(const v4f*)(p + 4 * j + (j >> 1) * 8);
#pragma unroll
    for (int i = 0; i < 4; ++i) f[4 * j + i] = (__bf16)v[i];
  }
  return f;
}

// ---------------------------------------------------------------------------
// Kernel 1: Qp = gelu(Q @ W^T + bias); store Qp [b][q][h] and QpT [b][h][q].
// ---------------------------------------------------------------------------
__global__ __launch_bounds__(32)
void linear_gelu_k(const float* __restrict__ Q, const float* __restrict__ W,
                   const float* __restrict__ bias,
                   __bf16* __restrict__ Qp, __bf16* __restrict__ QpT) {
  const int lane = threadIdx.x & 31;
  const int half = lane >> 4;
  const int lm   = lane & 15;
  const int blk  = blockIdx.x;
  const int ot = blk & 15;            // H/16  = 16
  const int qt = (blk >> 4) & 127;    // QN/16 = 128
  const int b  = blk >> 11;
  const int q0 = qt * 16, o0 = ot * 16;

  const float* Qb = Q + (size_t)b * QN * H_;
  v8f acc = {};
#pragma unroll
  for (int ks = 0; ks < 8; ++ks) {
    v16bf a  = frag_a_from_f32(Qb + (size_t)(q0 + lm) * H_ + ks * 32 + half * 8);
    v16bf bf = frag_b_from_f32(W  + (size_t)(o0 + lm) * H_ + ks * 32 + half * 8);
    acc = wmma_bf16(a, bf, acc);
  }
  const float bv = bias[o0 + lm];
  v8bf pk;
  const size_t qpo = (size_t)b * QN * H_;
#pragma unroll
  for (int r = 0; r < 8; ++r) {
    float x = acc[r] + bv;
    float g = 0.5f * x * (1.0f + erff(x * 0.70710678118654752f));  // exact GELU
    __bf16 gb = (__bf16)g;
    int m = r + half * 8;                        // D row m = q-local
    Qp[qpo + (size_t)(q0 + m) * H_ + (o0 + lm)] = gb;
    pk[r] = gb;                                  // q consecutive within lane
  }
  *(v8bf*)(QpT + (size_t)b * H_ * QN + (size_t)(o0 + lm) * QN + (q0 + half * 8)) = pk;
}

// ---------------------------------------------------------------------------
// Kernel 2: per (batch, 16 context rows): online-softmax attention over q.
// ---------------------------------------------------------------------------
__global__ __launch_bounds__(32)
void attn_k(const float* __restrict__ C, const __bf16* __restrict__ Qp,
            const __bf16* __restrict__ QpT, float* __restrict__ out) {
  __shared__ __align__(32) __bf16 ldsC[8 * 32 * 16];        // cached C A-frags (8 KB)
  __shared__ __align__(32) __bf16 ldsP[16 * LP_PITCH];      // P tile d x 64q (2.25 KB)
#if HAS_ASYNC
  __shared__ __align__(32) __bf16 qbuf[2][CH * QB_PITCH];   // staged Qp chunks (2x33 KB)
#endif

  const int lane = threadIdx.x & 31;
  const int half = lane >> 4;
  const int lm   = lane & 15;
  const int blk  = blockIdx.x;
  const int dt = blk & 511;           // DN/16 = 512
  const int b  = blk >> 9;
  const int d0 = dt * 16;

  const float*  Cb   = C   + (size_t)b * DN * H_;
  const __bf16* Qpb  = Qp  + (size_t)b * QN * H_;
  const __bf16* QpTb = QpT + (size_t)b * H_ * QN;

  // Cache the loop-invariant C-tile A-fragments (f32 -> bf16) in LDS.
#pragma unroll
  for (int ks = 0; ks < 8; ++ks) {
    v16bf a = frag_a_from_f32(Cb + (size_t)(d0 + lm) * H_ + ks * 32 + half * 8);
    v8bf lo, hi;
#pragma unroll
    for (int i = 0; i < 8; ++i) { lo[i] = a[i]; hi[i] = a[8 + i]; }
    v8bf* dst = (v8bf*)(ldsC + (ks * 32 + lane) * 16);
    dst[0] = lo; dst[1] = hi;
  }
  __syncthreads();

#if HAS_ASYNC
  // Stage q-chunk [q0g..q0g+63] x H of Qp into LDS: one async b128 per row
  // (32 lanes x 16B = 512B row). 64 issues per chunk; ASYNCcnt completes
  // in order, so issuing the next chunk's 64 loads drains the previous one.
  auto stage_chunk = [&](int q0g, __bf16* lbase) {
#pragma unroll
    for (int q = 0; q < CH; ++q) {
      const __bf16* g = Qpb + (size_t)(q0g + q) * H_ + lane * 8;
      __bf16* l = lbase + q * QB_PITCH + lane * 8;
      __builtin_amdgcn_global_load_async_to_lds_b128(
          (gas_v4i*)g, (las_v4i*)l, /*offset=*/0, /*cpol=*/0);
    }
  };
  stage_chunk(0, &qbuf[0][0]);
#endif

  float m_i[8], l_i[8];
  v8f acc[16];
  const v8f vzero = {};
#pragma unroll
  for (int r = 0; r < 8; ++r) { m_i[r] = -3.0e38f; l_i[r] = 0.0f; }
#pragma unroll
  for (int t = 0; t < 16; ++t) acc[t] = vzero;

  for (int q0 = 0; q0 < QN; q0 += CH) {
#if HAS_ASYNC
    const int cur = (q0 >> 6) & 1;
    if (q0 + CH < QN) stage_chunk(q0 + CH, &qbuf[cur ^ 1][0]);
    else              wait_async_all();
    const __bf16* qsrc = &qbuf[cur][0];
    const int qpitch = QB_PITCH;
#else
    const __bf16* qsrc = Qpb + (size_t)q0 * H_;
    const int qpitch = H_;
#endif
    __syncthreads();

    // ---- logits: four 16x16 tiles (64 q) over K = 256 ---------------------
    v8f S0 = {}, S1 = {}, S2 = {}, S3 = {};
#pragma unroll
    for (int ks = 0; ks < 8; ++ks) {
      v16bf a  = *(const v16bf*)(ldsC + (ks * 32 + lane) * 16);
      v16bf b0 = frag_b_from_bf16(qsrc + (size_t)(0  + lm) * qpitch + ks * 32 + half * 8);
      v16bf b1 = frag_b_from_bf16(qsrc + (size_t)(16 + lm) * qpitch + ks * 32 + half * 8);
      v16bf b2 = frag_b_from_bf16(qsrc + (size_t)(32 + lm) * qpitch + ks * 32 + half * 8);
      v16bf b3 = frag_b_from_bf16(qsrc + (size_t)(48 + lm) * qpitch + ks * 32 + half * 8);
      S0 = wmma_bf16(a, b0, S0);      // 4 independent chains hide WMMA RAW hazard
      S1 = wmma_bf16(a, b1, S1);
      S2 = wmma_bf16(a, b2, S2);
      S3 = wmma_bf16(a, b3, S3);
    }

    // ---- online softmax over the 64 new q columns -------------------------
#pragma unroll
    for (int r = 0; r < 8; ++r) {
      float mx = fmaxf(fmaxf(S0[r], S1[r]), fmaxf(S2[r], S3[r]));
#pragma unroll
      for (int msk = 1; msk <= 8; msk <<= 1) mx = fmaxf(mx, __shfl_xor(mx, msk, 32));
      float newm = fmaxf(m_i[r], mx);
      float p0 = __expf(S0[r] - newm);
      float p1 = __expf(S1[r] - newm);
      float p2 = __expf(S2[r] - newm);
      float p3 = __expf(S3[r] - newm);
      float rs = (p0 + p1) + (p2 + p3);
#pragma unroll
      for (int msk = 1; msk <= 8; msk <<= 1) rs += __shfl_xor(rs, msk, 32);
      float corr = __expf(m_i[r] - newm);
      l_i[r] = l_i[r] * corr + rs;
      m_i[r] = newm;
#pragma unroll
      for (int t = 0; t < 16; ++t) acc[t][r] *= corr;
      int m = r + half * 8;
      ldsP[m * LP_PITCH + 0  + lm] = (__bf16)p0;
      ldsP[m * LP_PITCH + 16 + lm] = (__bf16)p1;
      ldsP[m * LP_PITCH + 32 + lm] = (__bf16)p2;
      ldsP[m * LP_PITCH + 48 + lm] = (__bf16)p3;
    }
    __syncthreads();

    // P as two A-fragments (16 x 32 each, K = q-local 0..31 / 32..63).
    v16bf pa0 = frag_a_from_bf16(ldsP + lm * LP_PITCH + 0  + half * 8);
    v16bf pa1 = frag_a_from_bf16(ldsP + lm * LP_PITCH + 32 + half * 8);

    if (q0 + CH < QN)   // gfx1250 global_prefetch_b8 hint for QpT stream
      __builtin_prefetch(QpTb + (size_t)(lane * 8) * QN + q0 + CH, 0, 3);

    // ---- attended += P @ Qp  (B from QpT: contiguous in K=q) --------------
#pragma unroll
    for (int t = 0; t < 16; ++t) {
      const __bf16* bp = QpTb + (size_t)(t * 16 + lm) * QN + q0;
      v16bf bf0 = frag_b_from_bf16(bp + 0  + half * 8);
      v16bf bf1 = frag_b_from_bf16(bp + 32 + half * 8);
      acc[t] = wmma_bf16(pa0, bf0, acc[t]);
      acc[t] = wmma_bf16(pa1, bf1, acc[t]);
    }
    __syncthreads();
  }

  // ---- epilogue: out = acc / l + C ----------------------------------------
  float* outb = out + (size_t)b * DN * H_;
#pragma unroll
  for (int t = 0; t < 16; ++t) {
#pragma unroll
    for (int r = 0; r < 8; ++r) {
      int m = r + half * 8;
      int n = t * 16 + lm;
      size_t idx = (size_t)(d0 + m) * H_ + n;
      outb[idx] = acc[t][r] / l_i[r] + Cb[idx];
    }
  }
}

// ---------------------------------------------------------------------------
extern "C" void kernel_launch(void* const* d_in, const int* in_sizes, int n_in,
                              void* d_out, int out_size, void* d_ws, size_t ws_size,
                              hipStream_t stream) {
  const float* Q  = (const float*)d_in[0];
  const float* C  = (const float*)d_in[1];
  const float* W  = (const float*)d_in[2];
  const float* bv = (const float*)d_in[3];
  float* out = (float*)d_out;

  // Workspace: Qp (bf16, [B][Q][H]) then QpT (bf16, [B][H][Q]) = 2 * 8.39 MB.
  __bf16* Qp  = (__bf16*)d_ws;
  __bf16* QpT = (__bf16*)((char*)d_ws + (size_t)B_ * QN * H_ * sizeof(__bf16));

  linear_gelu_k<<<B_ * (QN / 16) * (H_ / 16), 32, 0, stream>>>(Q, W, bv, Qp, QpT);
  attn_k<<<B_ * (DN / 16), 32, 0, stream>>>(C, Qp, QpT, out);
}
